// Shader_36971078484036
// MI455X (gfx1250) — matrix operand order repaired and checked by reading, not compiled, not run on hardware
//
#include <hip/hip_runtime.h>
#include <hip/hip_bf16.h>

typedef __attribute__((ext_vector_type(16))) __bf16 v16bf;
typedef __attribute__((ext_vector_type(8)))  float  v8f;
typedef __attribute__((ext_vector_type(4)))  unsigned int u32x4;
typedef __attribute__((ext_vector_type(8)))  int  i32x8;
typedef __attribute__((ext_vector_type(4)))  int  i32x4;

// Problem constants
#define BB   32
#define CC   384
#define HH   32
#define WW   32
#define KK   64
#define HWX  1024
#define HP   34           // padded H/W
#define NPIX (BB*HH*WW)   // 32768

static constexpr int PK_ELEMS  = 9 * 12 * 24 * 32 * 16;     // 1,327,104 bf16
static constexpr int PAD_ELEMS = BB * HP * HP * CC;         // 14,204,928 bf16

// LDS tile: [3 rows][34 cols][32 cin + 8 pad] bf16 = 4080 elems = 8160 B
static constexpr int LROW = 40;                 // padded cin stride (elems)
static constexpr int LTILE = 3 * 34 * LROW;     // 4080 elems

// Workspace byte offsets (256-aligned)
static constexpr size_t WS_NMEAN = 0;                       // [2][32][384] f32
static constexpr size_t WS_NSTD  = 98304;                   // [2][32][384] f32
static constexpr size_t WS_BNP   = 196608;                  // [384][2] f32 (scale,shift)
static constexpr size_t WS_PK    = 199680;                  // packed weights bf16
static constexpr size_t WS_ACT   = 2853888;                 // padded NHWC bf16 activations

// ---------------------------------------------------------------------------
// K0: zero the padded activation buffer (halo must be 0; ws is poisoned)
__global__ __launch_bounds__(256) void zero_pad(uint4* __restrict__ p, int n16) {
    for (int i = blockIdx.x * blockDim.x + threadIdx.x; i < n16;
         i += gridDim.x * blockDim.x)
        p[i] = make_uint4(0u, 0u, 0u, 0u);
}

// ---------------------------------------------------------------------------
// K1: newMean/newStd[l][b][c] = w[l,b,:] @ proto_{pos|neg}[l,:,c]  (label select)
__global__ __launch_bounds__(256) void affine_tables(
    const float* __restrict__ pmp, const float* __restrict__ psp,
    const float* __restrict__ pmn, const float* __restrict__ psn,
    const float* __restrict__ cw,  const int*   __restrict__ label,
    float* __restrict__ nmean, float* __restrict__ nstd) {
    int i = blockIdx.x * 256 + threadIdx.x;        // i = (l*32+b)*384 + c
    if (i >= 2 * BB * CC) return;
    int c = i % CC;
    int t = i / CC;
    int b = t % BB;
    int l = t / BB;
    const int lbl = label[b];
    const float* pm = (lbl == 0) ? pmp : pmn;
    const float* ps = (lbl == 0) ? psp : psn;
    const float* w  = cw + (l * BB + b) * KK;
    float am = 0.f, as = 0.f;
    #pragma unroll 4
    for (int k = 0; k < KK; ++k) {
        const float wk = w[k];
        am = fmaf(wk, pm[(l * KK + k) * CC + c], am);
        as = fmaf(wk, ps[(l * KK + k) * CC + c], as);
    }
    nmean[i] = am;
    nstd[i]  = as;
}

// ---------------------------------------------------------------------------
// K2: fused double-CSA per (b,c) plane; writes bf16 into padded NHWC layout
__global__ __launch_bounds__(256) void csa_fused(
    const float* __restrict__ x, const float* __restrict__ nmean,
    const float* __restrict__ nstd, __bf16* __restrict__ actp) {
    __shared__ float rs[256];
    __shared__ float rss[256];
    const int tid = threadIdx.x;
    const int bid = blockIdx.x;                    // (b*384 + c)
    const int b   = bid / CC;
    const int c   = bid - b * CC;

    float4 v = ((const float4*)x)[bid * 256 + tid];

    // --- reduction 1 (sum, sumsq) ---
    float s  = v.x + v.y + v.z + v.w;
    float ss = fmaf(v.x, v.x, fmaf(v.y, v.y, fmaf(v.z, v.z, v.w * v.w)));
    rs[tid] = s; rss[tid] = ss;
    __syncthreads();
    for (int off = 128; off > 0; off >>= 1) {
        if (tid < off) { rs[tid] += rs[tid + off]; rss[tid] += rss[tid + off]; }
        __syncthreads();
    }
    float sum = rs[0], sumsq = rss[0];
    __syncthreads();

    const int base = b * CC + c;
    float mean = sum * (1.f / 1024.f);
    float var  = (sumsq - 1024.f * mean * mean) * (1.f / 1023.f);  // ddof=1
    float inv  = rsqrtf(var + 1e-5f);
    float s1 = nstd[base], m1 = nmean[base];
    float a1 = 1.f + s1 * inv;
    float d1 = m1 - mean * s1 * inv;                // fea1 = x*a1 + d1
    float4 f;
    f.x = fmaf(v.x, a1, d1); f.y = fmaf(v.y, a1, d1);
    f.z = fmaf(v.z, a1, d1); f.w = fmaf(v.w, a1, d1);

    // --- reduction 2 ---
    s  = f.x + f.y + f.z + f.w;
    ss = fmaf(f.x, f.x, fmaf(f.y, f.y, fmaf(f.z, f.z, f.w * f.w)));
    rs[tid] = s; rss[tid] = ss;
    __syncthreads();
    for (int off = 128; off > 0; off >>= 1) {
        if (tid < off) { rs[tid] += rs[tid + off]; rss[tid] += rss[tid + off]; }
        __syncthreads();
    }
    sum = rs[0]; sumsq = rss[0];

    mean = sum * (1.f / 1024.f);
    var  = (sumsq - 1024.f * mean * mean) * (1.f / 1023.f);
    inv  = rsqrtf(var + 1e-5f);
    float s2 = nstd[12288 + base], m2 = nmean[12288 + base];
    float a2 = 1.f + s2 * inv;
    float d2 = m2 - mean * s2 * inv;
    float4 g;
    g.x = fmaf(f.x, a2, d2); g.y = fmaf(f.y, a2, d2);
    g.z = fmaf(f.z, a2, d2); g.w = fmaf(f.w, a2, d2);

    // store bf16 into padded NHWC: [b][h+1][w+1][c]
    const int p0 = tid * 4;
    const int h  = p0 >> 5;
    const int w  = p0 & 31;                        // 4 pixels never cross a row
    __bf16* d = actp + (((size_t)b * HP + (h + 1)) * HP + (w + 1)) * CC + c;
    d[0 * CC] = (__bf16)g.x;
    d[1 * CC] = (__bf16)g.y;
    d[2 * CC] = (__bf16)g.z;
    d[3 * CC] = (__bf16)g.w;
}

// ---------------------------------------------------------------------------
// K3: pack conv_w [co][ci][3][3] f32 -> bf16 WMMA B-fragment order
//     pk[tap][kb][nb][lane(32)][e(16)]: co = lane%16 (+k-half lane/16), e indexes ci
__global__ __launch_bounds__(256) void pack_weights(
    const float* __restrict__ cw, __bf16* __restrict__ pk) {
    int i = blockIdx.x * 256 + threadIdx.x;
    if (i >= PK_ELEMS) return;
    int e    = i & 15;  int t = i >> 4;
    int lanek = t & 31; t >>= 5;
    int nb   = t % 24;  t /= 24;
    int kb   = t % 12;
    int tap  = t / 12;
    int co = nb * 16 + (lanek & 15);
    int ci = kb * 32 + (e & 7) + ((lanek >> 4) << 3) + ((e >> 3) << 4);
    pk[i] = (__bf16)cw[((size_t)co * CC + ci) * 9 + tap];
}

// ---------------------------------------------------------------------------
// K4: implicit-GEMM 3x3 conv via v_wmma_f32_16x16x32_bf16 + TDM halo staging.
// Grid: (3 cout-blocks-of-128, B*H rows). WG = 256 = 8 waves; each wave owns a
// 16-cout block and both 16-pixel halves of the output row (2 accumulators).
__global__ __launch_bounds__(256) void conv_wmma(
    const __bf16* __restrict__ actp, const __bf16* __restrict__ pk,
    float* __restrict__ y) {
    __shared__ __bf16 sA[2][LTILE];                // double-buffered padded tile

    const int tid  = threadIdx.x;
    const int lane = tid & 31;
    const int wave = tid >> 5;
    const int b    = blockIdx.y >> 5;
    const int h    = blockIdx.y & 31;
    const int nb   = blockIdx.x * 8 + wave;        // global cout/16 block
    const int lw   = lane & 15;
    const int lo   = (lane < 16) ? 0 : 8;          // A-fragment K-half select

    v8f acc0 = {0.f, 0.f, 0.f, 0.f, 0.f, 0.f, 0.f, 0.f};
    v8f acc1 = {0.f, 0.f, 0.f, 0.f, 0.f, 0.f, 0.f, 0.f};

    // --- Tensor DMA descriptor (ISA ch.8): 3D tile {cin32, col34, row3} ---
    // group1: data_size=2B(code1), pad_enable, pad_interval=16DW(code3),
    //         pad_amount=4DW(code3) -> per-(row,col) LDS stride = 40 bf16
    i32x8 g1;
    g1[0] = 0x00010000 | (1 << 20) | (3 << 22) | (3 << 25);
    g1[1] = 32 << 16;                // tensor_dim0[15:0]=32   (bits 79:48)
    g1[2] = 34 << 16;                // tensor_dim1[15:0]=34   (bits 111:80)
    g1[3] = 32 << 16;                // tile_dim0=32           (bits 127:112)
    g1[4] = 34 | (3 << 16);          // tile_dim1=34, tile_dim2=3
    g1[5] = 384;                     // tensor_dim0_stride (elems)
    g1[6] = (int)(0x3300u << 16);    // tensor_dim1_stride=13056 (bits 255:208)
    g1[7] = 0;
    i32x4 g2; g2[0] = 3; g2[1] = 0; g2[2] = 0; g2[3] = 0;   // tensor_dim2=3
    i32x4 g3; g3[0] = 0; g3[1] = 0; g3[2] = 0; g3[3] = 0;
    i32x8 g4;                                                // unused extra words
    g4[0] = 0; g4[1] = 0; g4[2] = 0; g4[3] = 0;
    g4[4] = 0; g4[5] = 0; g4[6] = 0; g4[7] = 0;

    const unsigned long long gbase = (unsigned long long)(uintptr_t)actp +
        2ull * (((unsigned long long)b * HP + h) * HP * CC);
    const unsigned int lds0 = (unsigned int)(uintptr_t)(void*)&sA[0][0];

    #define TDM_ISSUE(kb_, buf_)                                               \
        do {                                                                   \
            unsigned long long ga_ = gbase + 2ull * ((kb_) * 32);              \
            u32x4 g0_;                                                         \
            g0_.x = 1u;                                  /* count=1 */         \
            g0_.y = lds0 + (buf_) * (LTILE * 2);         /* lds_addr */        \
            g0_.z = (unsigned int)(ga_ & 0xffffffffu);                         \
            g0_.w = (unsigned int)((ga_ >> 32) & 0x1ffffffu) | (2u << 30);     \
            __builtin_amdgcn_tensor_load_to_lds(g0_, g1, g2, g3, g4, 0);       \
        } while (0)

    if (wave == 0) TDM_ISSUE(0, 0);

    for (int kb = 0; kb < 12; ++kb) {
        if (wave == 0) __builtin_amdgcn_s_wait_tensorcnt(0);
        __syncthreads();                           // tile kb is ready for all
        if (wave == 0 && kb < 11) TDM_ISSUE(kb + 1, (kb + 1) & 1);

        const __bf16* buf = &sA[kb & 1][0];
        #pragma unroll
        for (int kh = 0; kh < 3; ++kh) {
            #pragma unroll
            for (int kw = 0; kw < 3; ++kw) {
                const int tap = kh * 3 + kw;
                // B: 32 cin x 16 cout, per-lane contiguous 32B (coalesced)
                union { uint4 u[2]; v16bf v; } ub;
                const uint4* wp = (const uint4*)(pk +
                    ((size_t)(((tap * 12 + kb) * 24 + nb) * 32 + lane) << 4));
                ub.u[0] = wp[0];
                ub.u[1] = wp[1];
                // A: per-lane 2x16B from padded LDS (conflict-free, 20-bank step)
                const int e0 = (kh * 34 + (lw + kw)) * LROW + lo;
                union { uint4 u[2]; v16bf v; } ua0, ua1;
                ua0.u[0] = *(const uint4*)(buf + e0);
                ua0.u[1] = *(const uint4*)(buf + e0 + 16);
                ua1.u[0] = *(const uint4*)(buf + e0 + 16 * LROW);
                ua1.u[1] = *(const uint4*)(buf + e0 + 16 * LROW + 16);
                acc0 = __builtin_amdgcn_wmma_f32_16x16x32_bf16(
                    false, ua0.v, false, ub.v, (short)0, acc0, false, false);
                acc1 = __builtin_amdgcn_wmma_f32_16x16x32_bf16(
                    false, ua1.v, false, ub.v, (short)0, acc1, false, false);
            }
        }
        __syncthreads();                           // reads done before reuse
    }
    #undef TDM_ISSUE

    // D layout: lanes 0-15: N=lane, M=g ; lanes 16-31: N=lane-16, M=g+8
    const int cout  = nb * 16 + lw;
    const int wbase = (lane < 16) ? 0 : 8;
    float* dst = y + (((size_t)b * CC + cout) * HWX + h * WW + wbase);
    #pragma unroll
    for (int gi = 0; gi < 8; ++gi) dst[gi]      = acc0[gi];
    #pragma unroll
    for (int gi = 0; gi < 8; ++gi) dst[16 + gi] = acc1[gi];
}

// ---------------------------------------------------------------------------
// K5: per-channel BN stats (biased var) folded into scale/shift
__global__ __launch_bounds__(256) void bn_stats(
    const float* __restrict__ y, const float* __restrict__ gamma,
    const float* __restrict__ beta, float* __restrict__ bnp) {
    __shared__ float rs[256];
    __shared__ float rss[256];
    const int c = blockIdx.x, tid = threadIdx.x;
    float s = 0.f, ss = 0.f;
    for (int p = tid; p < NPIX; p += 256) {
        const float v = y[(((size_t)(p >> 10) * CC + c) << 10) + (p & 1023)];
        s += v;
        ss = fmaf(v, v, ss);
    }
    rs[tid] = s; rss[tid] = ss;
    __syncthreads();
    for (int off = 128; off > 0; off >>= 1) {
        if (tid < off) { rs[tid] += rs[tid + off]; rss[tid] += rss[tid + off]; }
        __syncthreads();
    }
    if (tid == 0) {
        const float mu  = rs[0] * (1.f / (float)NPIX);
        const float var = rss[0] * (1.f / (float)NPIX) - mu * mu;
        const float sc  = gamma[c] * rsqrtf(var + 1e-5f);
        bnp[2 * c]     = sc;
        bnp[2 * c + 1] = beta[c] - mu * sc;
    }
}

// K6: in-place BN apply on d_out
__global__ __launch_bounds__(256) void bn_apply(
    float* __restrict__ y, const float* __restrict__ bnp) {
    const int n4 = (BB * CC * HWX) / 4;            // 3,145,728
    for (int i = blockIdx.x * blockDim.x + threadIdx.x; i < n4;
         i += gridDim.x * blockDim.x) {
        const int c  = (i >> 8) % 384;
        const float sc = bnp[2 * c], sh = bnp[2 * c + 1];
        float4 v = ((float4*)y)[i];
        v.x = fmaf(v.x, sc, sh); v.y = fmaf(v.y, sc, sh);
        v.z = fmaf(v.z, sc, sh); v.w = fmaf(v.w, sc, sh);
        ((float4*)y)[i] = v;
    }
}

// ---------------------------------------------------------------------------
extern "C" void kernel_launch(void* const* d_in, const int* in_sizes, int n_in,
                              void* d_out, int out_size, void* d_ws, size_t ws_size,
                              hipStream_t stream) {
    const float* x     = (const float*)d_in[0];
    const float* pmp   = (const float*)d_in[1];
    const float* psp   = (const float*)d_in[2];
    const float* pmn   = (const float*)d_in[3];
    const float* psn   = (const float*)d_in[4];
    const float* cw    = (const float*)d_in[5];
    const float* convw = (const float*)d_in[6];
    const float* gamma = (const float*)d_in[7];
    const float* beta  = (const float*)d_in[8];
    const int*   label = (const int*)d_in[9];
    float* out = (float*)d_out;

    char* ws = (char*)d_ws;
    float*  nmean = (float*)(ws + WS_NMEAN);
    float*  nstd  = (float*)(ws + WS_NSTD);
    float*  bnp   = (float*)(ws + WS_BNP);
    __bf16* pk    = (__bf16*)(ws + WS_PK);
    __bf16* actp  = (__bf16*)(ws + WS_ACT);

    // K0: zero padded activation buffer (halo)
    zero_pad<<<2048, 256, 0, stream>>>((uint4*)actp, PAD_ELEMS / 8);
    // K1: affine tables (folds label select + w@proto)
    affine_tables<<<(2 * BB * CC + 255) / 256, 256, 0, stream>>>(
        pmp, psp, pmn, psn, cw, label, nmean, nstd);
    // K3: weight pack (independent of K0..K2)
    pack_weights<<<(PK_ELEMS + 255) / 256, 256, 0, stream>>>(convw, pk);
    // K2: fused double CSA -> bf16 padded NHWC
    csa_fused<<<BB * CC, 256, 0, stream>>>(x, nmean, nstd, actp);
    // K4: WMMA conv (TDM-staged halo) -> d_out (f32, NCHW)
    conv_wmma<<<dim3(3, BB * HH), 256, 0, stream>>>(actp, pk, out);
    // K5: BN stats -> scale/shift
    bn_stats<<<CC, 256, 0, stream>>>(out, gamma, beta, bnp);
    // K6: BN apply in place
    bn_apply<<<3072, 256, 0, stream>>>(out, bnp);
}
